// MyModel_60344290508853
// MI455X (gfx1250) — matrix operand order repaired and checked by reading, not compiled, not run on hardware
//
#include <hip/hip_runtime.h>
#include <hip/hip_bf16.h>
#include <math.h>
#include <stdint.h>

// ---------------- CDNA5 vector types ----------------
typedef __attribute__((ext_vector_type(16))) __bf16 v16bf;
typedef __attribute__((ext_vector_type(8)))  __bf16 v8bf;
typedef __attribute__((ext_vector_type(8)))  float  v8f;
typedef __attribute__((ext_vector_type(4)))  int    v4i;

#define VOCAB 32000
#define EMB   128
#define HID   256
#define G3    768      // 3*HID
#define BATCH 512
#define TLEN  512

// gfx1250 async global->LDS DMA path (ASYNCcnt), if the toolchain exposes it.
#if __has_builtin(__builtin_amdgcn_global_load_async_to_lds_b128) && \
    __has_builtin(__builtin_amdgcn_s_wait_asynccnt)
#define HAVE_ASYNC 1
#else
#define HAVE_ASYNC 0
#endif

#if HAVE_ASYNC
typedef __attribute__((address_space(1))) v4i* as1_v4i;   // global int4*
typedef __attribute__((address_space(3))) v4i* as3_v4i;   // LDS int4*
__device__ __forceinline__ as1_v4i to_global(const void* p) {
    return (as1_v4i)(uintptr_t)p;                 // global generic == AS1 value
}
__device__ __forceinline__ as3_v4i to_lds(const void* p) {
    return (as3_v4i)(uint32_t)(uintptr_t)p;       // low 32 bits == LDS offset
}
#endif

// ---------------------------------------------------------------------------
// Kernel 0: fp32 -> bf16 conversion of emb / W_ih / W_hh into workspace.
// ---------------------------------------------------------------------------
__global__ void gru_cvt_bf16(const float* __restrict__ emb,
                             const float* __restrict__ wih,
                             const float* __restrict__ whh,
                             __bf16* __restrict__ emb16,
                             __bf16* __restrict__ wih16,
                             __bf16* __restrict__ whh16) {
    int i = blockIdx.x * blockDim.x + threadIdx.x;
    int stride = gridDim.x * blockDim.x;
    for (int j = i; j < VOCAB * EMB; j += stride) emb16[j] = (__bf16)emb[j];
    for (int j = i; j < G3 * EMB;    j += stride) wih16[j] = (__bf16)wih[j];
    for (int j = i; j < G3 * HID;    j += stride) whh16[j] = (__bf16)whh[j];
}

// ---------------------------------------------------------------------------
// Kernel 1: gi[t,b,g] = emb[x[b,t]] @ W_ih^T + b_ih   (fp32 out, bf16 WMMA)
// Grid: (T*B)/16 workgroups of 256 threads (8 waves).
// ---------------------------------------------------------------------------
__global__ __launch_bounds__(256)
void gru_gi_gemm(const int*    __restrict__ x,
                 const __bf16* __restrict__ emb16,
                 const __bf16* __restrict__ wih16,
                 const float*  __restrict__ b_ih,
                 float*        __restrict__ gi) {
    __shared__ __align__(16) __bf16 Atile[16 * EMB];   // 4 KB bf16 A tile

    const int  tid = threadIdx.x;
    const long r0  = (long)blockIdx.x * 16;            // global row base (t*B+b)

    // Stage A: gather 16 embedded token rows (bf16, 256B each) into LDS.
    {
        int  row = tid >> 4;                 // 0..15
        int  c16 = tid & 15;                 // 16B chunk within row
        long r   = r0 + row;
        int  t   = (int)(r / BATCH);
        int  b   = (int)(r % BATCH);
        int  tok = x[(long)b * TLEN + t];
        const uint4* src = (const uint4*)(emb16 + (long)tok * EMB);
        uint4*       dst = (uint4*)(Atile + row * EMB);
        dst[c16] = src[c16];
    }
    __syncthreads();

    const int wv   = tid >> 5;      // wave 0..7
    const int lane = tid & 31;
    const int lhi  = lane >> 4;     // 0|1
    const int lmod = lane & 15;

    for (int j = 0; j < 6; ++j) {
        const int n0   = wv * 96 + 16 * j;              // N-tile base (0..752)
        const float bv = b_ih[n0 + lmod];               // bias folded into C
        v8f acc = {bv, bv, bv, bv, bv, bv, bv, bv};

        for (int k = 0; k < 4; ++k) {                   // K = 128 = 4 x 32
            const int kb = k * 32 + (lhi ? 16 : 0);
            const v8bf* ap  = (const v8bf*)(Atile + lmod * EMB + kb);
            v8bf alo = ap[0];                           // K kb..kb+7
            v8bf ahi = ap[1];                           // K kb+8..kb+15
            v16bf a = __builtin_shufflevector(alo, ahi,
                      0,1,2,3,4,5,6,7,8,9,10,11,12,13,14,15);
            const v16bf* bp = (const v16bf*)
                (wih16 + (long)(n0 + lmod) * EMB + k * 32 + lhi * 16);
            acc = __builtin_amdgcn_wmma_f32_16x16x32_bf16(
                      false, a, false, *bp, (short)0, acc, false, false);
        }

        float* gout = gi + r0 * G3;
        #pragma unroll
        for (int v = 0; v < 8; ++v) {
            int m = v + 8 * lhi;
            gout[(long)m * G3 + n0 + lmod] = acc[v];
        }
    }
}

// ---------------------------------------------------------------------------
// Kernel 2: sequential GRU scan. 32 WGs x 512 threads (16 waves).
// WG owns 16 batch rows. Wave w owns cols [16w,16w+16) of EVERY gate, so the
// r/z/n accumulators are elementwise-aligned in registers. W_hh B-fragments
// are register-resident across the whole 512-step scan (24 x v16bf = 192
// VGPRs; gfx1250 waves address up to 1024 VGPRs). h lives in LDS as bf16
// (A-frags = two ds_load_b128), while the fp32 h_prev for the z-blend stays
// in each lane's registers. gi slabs are double-buffered in LDS via async
// global->LDS DMA, overlapped with the WMMA K-loop.
// ---------------------------------------------------------------------------
#define HSTRB (HID + 8)          // bf16 h stride: 16B pad -> bank spread

__global__ __launch_bounds__(512)
void gru_scan(const float*  __restrict__ gi,
              const __bf16* __restrict__ whh16,
              const float*  __restrict__ b_hh,
              float*        __restrict__ out) {
    __shared__ __align__(16) __bf16 hs[16 * HSTRB];          // 8.25 KB
#if HAVE_ASYNC
    __shared__ __align__(16) float gibuf[2][16 * G3];        // 2 x 48 KB
#endif

    const int tid = threadIdx.x;
    const int b0  = blockIdx.x * 16;

    for (int i = tid; i < 16 * HSTRB; i += 512) hs[i] = (__bf16)0.f;

    const int wv   = tid >> 5;      // wave 0..15
    const int lane = tid & 31;
    const int lhi  = lane >> 4;
    const int lmod = lane & 15;
    const int cb   = wv * 16;       // column base within each gate
    const int col  = cb + lmod;

    // Register-resident W_hh B fragments: bw[gate][k]
    v16bf bw[3][8];
    #pragma unroll
    for (int g = 0; g < 3; ++g)
        #pragma unroll
        for (int k = 0; k < 8; ++k)
            bw[g][k] = *(const v16bf*)
                (whh16 + (long)(g * HID + col) * HID + k * 32 + lhi * 16);

    float bias[3];
    #pragma unroll
    for (int g = 0; g < 3; ++g) bias[g] = b_hh[g * HID + col];

    float hprev[8];
    #pragma unroll
    for (int v = 0; v < 8; ++v) hprev[v] = 0.f;

#if HAVE_ASYNC
    // Prime buffer 0 with gi slab for t=0: 16x768 fp32 = 3072 x 16B chunks.
    {
        const float* src = gi + (long)b0 * G3;
        #pragma unroll
        for (int q = 0; q < 6; ++q) {
            int c = tid + 512 * q;
            __builtin_amdgcn_global_load_async_to_lds_b128(
                to_global(src + c * 4), to_lds(&gibuf[0][c * 4]), 0, 0);
        }
    }
#endif
    __syncthreads();

    for (int t = 0; t < TLEN; ++t) {
#if HAVE_ASYNC
        if (t + 1 < TLEN) {     // prefetch next slab into the other buffer
            const float* src = gi + ((long)(t + 1) * BATCH + b0) * G3;
            float* dst = gibuf[(t + 1) & 1];
            #pragma unroll
            for (int q = 0; q < 6; ++q) {
                int c = tid + 512 * q;
                __builtin_amdgcn_global_load_async_to_lds_b128(
                    to_global(src + c * 4), to_lds(dst + c * 4), 0, 0);
            }
        }
#endif
        v8f acc[3];
        #pragma unroll
        for (int g = 0; g < 3; ++g) {
            float bv = bias[g];
            acc[g] = (v8f){bv, bv, bv, bv, bv, bv, bv, bv};
        }

        #pragma unroll
        for (int k = 0; k < 8; ++k) {            // K = 256 = 8 x 32
            const int kb = k * 32 + (lhi ? 16 : 0);
            const v8bf* ap = (const v8bf*)(hs + lmod * HSTRB + kb);
            v8bf alo = ap[0];
            v8bf ahi = ap[1];
            v16bf a = __builtin_shufflevector(alo, ahi,
                      0,1,2,3,4,5,6,7,8,9,10,11,12,13,14,15);
            #pragma unroll
            for (int g = 0; g < 3; ++g)
                acc[g] = __builtin_amdgcn_wmma_f32_16x16x32_bf16(
                    false, a, false, bw[g][k], (short)0, acc[g], false, false);
        }

#if HAVE_ASYNC
        if (t + 1 < TLEN) __builtin_amdgcn_s_wait_asynccnt(6);  // t's slab done
        else              __builtin_amdgcn_s_wait_asynccnt(0);
#else
        if (t + 1 < TLEN)
            __builtin_prefetch(gi + ((long)(t + 1) * BATCH + b0) * G3, 0, 0);
#endif
        __syncthreads();    // gi slab visible; all hs reads of step t complete

        float hnew[8];
        #pragma unroll
        for (int v = 0; v < 8; ++v) {
            const int m = v + 8 * lhi;
#if HAVE_ASYNC
            const float* grow = &gibuf[t & 1][m * G3];
#else
            const float* grow = gi + ((long)t * BATCH + b0 + m) * G3;
#endif
            float ir  = grow[col];
            float iz  = grow[HID + col];
            float in_ = grow[2 * HID + col];
            float rr  = 1.f / (1.f + __expf(-(ir + acc[0][v])));
            float zz  = 1.f / (1.f + __expf(-(iz + acc[1][v])));
            float nn  = tanhf(in_ + rr * acc[2][v]);
            hnew[v]   = (1.f - zz) * nn + zz * hprev[v];
        }

        #pragma unroll
        for (int v = 0; v < 8; ++v) {
            hs[(v + 8 * lhi) * HSTRB + col] = (__bf16)hnew[v];
            hprev[v] = hnew[v];
        }
        __syncthreads();    // h fully updated for step t+1
    }

    // Final h lives in this lane's registers; write straight to d_out.
    #pragma unroll
    for (int v = 0; v < 8; ++v)
        out[(long)(b0 + v + 8 * lhi) * HID + col] = hprev[v];
}

// ---------------------------------------------------------------------------
extern "C" void kernel_launch(void* const* d_in, const int* in_sizes, int n_in,
                              void* d_out, int out_size, void* d_ws, size_t ws_size,
                              hipStream_t stream) {
    (void)in_sizes; (void)n_in; (void)out_size; (void)ws_size;

    const int*   x    = (const int*)  d_in[0];   // (B,T) int32
    const float* emb  = (const float*)d_in[1];   // (VOCAB,EMB)
    const float* Wih  = (const float*)d_in[2];   // (3H,EMB)
    const float* Whh  = (const float*)d_in[3];   // (3H,HID)
    const float* b_ih = (const float*)d_in[4];   // (3H)
    const float* b_hh = (const float*)d_in[5];   // (3H)
    float*       out  = (float*)d_out;           // (B,HID)

    char* ws = (char*)d_ws;
    __bf16* emb16 = (__bf16*)(ws);                                // 8,192,000 B
    __bf16* wih16 = (__bf16*)(ws + 8192000);                      //   196,608 B
    __bf16* whh16 = (__bf16*)(ws + 8388608);                      //   393,216 B
    float*  gi    = (float*) (ws + 8781824);                      // 805,306,368 B

    gru_cvt_bf16<<<4096, 256, 0, stream>>>(emb, Wih, Whh, emb16, wih16, whh16);

    gru_gi_gemm<<<(TLEN * BATCH) / 16, 256, 0, stream>>>(x, emb16, wih16, b_ih, gi);

    gru_scan<<<BATCH / 16, 512, 0, stream>>>(gi, whh16, b_hh, out);
}
